// Attention_13838384628186
// MI455X (gfx1250) — matrix operand order repaired and checked by reading, not compiled, not run on hardware
//
#include <hip/hip_runtime.h>

// ---------------- problem constants (from reference) ----------------
#define S_LEN 2048
#define NB    2
#define HQ    32
#define HKV_N 8
#define DH    128
#define BM    128   // Q rows per block (8 waves x 16 rows)
#define KT    32    // keys per tile
#define KFP   132   // padded f32 row (128 + 4) -> conflict-free, 16B aligned

typedef __attribute__((ext_vector_type(16))) __bf16 v16bf;
typedef __attribute__((ext_vector_type(8)))  __bf16 v8bf;
typedef __attribute__((ext_vector_type(8)))  float  v8f;

union Frag16 {
  v16bf v;
  v8bf  h8[2];
};

__device__ __forceinline__ float red_max16(float x) {
  x = fmaxf(x, __shfl_xor(x, 1, 32));
  x = fmaxf(x, __shfl_xor(x, 2, 32));
  x = fmaxf(x, __shfl_xor(x, 4, 32));
  x = fmaxf(x, __shfl_xor(x, 8, 32));
  return x;
}
__device__ __forceinline__ float red_sum16(float x) {
  x += __shfl_xor(x, 1, 32);
  x += __shfl_xor(x, 2, 32);
  x += __shfl_xor(x, 4, 32);
  x += __shfl_xor(x, 8, 32);
  return x;
}

// ---- CDNA5 async Global->LDS copy (ASYNCcnt path), inline asm: portable ----
__device__ __forceinline__ void async_ld16(const void* lds, const float* g) {
  asm volatile("global_load_async_to_lds_b128 %0, %1, off"
               :: "v"((unsigned)(uintptr_t)lds), "v"(g)
               : "memory");
}
__device__ __forceinline__ void wait_async0() {
  asm volatile("s_wait_asynccnt 0x0" ::: "memory");
}

__device__ __forceinline__ v8bf cvt_f32x8(const float4 a, const float4 b) {
  v8bf r;
  r[0] = (__bf16)a.x; r[1] = (__bf16)a.y; r[2] = (__bf16)a.z; r[3] = (__bf16)a.w;
  r[4] = (__bf16)b.x; r[5] = (__bf16)b.y; r[6] = (__bf16)b.z; r[7] = (__bf16)b.w;
  return r;
}

// ---------------- KV-cache scatter store ----------------
__global__ __launch_bounds__(256) void kv_store_kernel(
    const float* __restrict__ k, const float* __restrict__ v,
    const int* __restrict__ slot, float* __restrict__ kc, float* __restrict__ vc) {
  const int row = blockIdx.x;          // 0 .. B*S-1
  const int t   = threadIdx.x;         // 256 threads, 1024 floats = 256 float4
  const int s   = slot[row];
  const float4* ks = (const float4*)k + (size_t)row * 256;
  const float4* vs = (const float4*)v + (size_t)row * 256;
  float4* kd = (float4*)kc + (size_t)s * 256;
  float4* vd = (float4*)vc + (size_t)s * 256;
  kd[t] = ks[t];
  vd[t] = vs[t];
}

// ------- causal GQA flash attention: async-pipelined, bf16 WMMA, f32 acc ----
__global__ __launch_bounds__(256) void fa_fwd(const float* __restrict__ q,
                                              const float* __restrict__ kg,
                                              const float* __restrict__ vg,
                                              float* __restrict__ out) {
  __shared__ __align__(16) float  Kf[KT][KFP];           // async staging (f32)
  __shared__ __align__(16) float  Vf[KT][KFP];           // async staging (f32)
  __shared__ __align__(16) __bf16 Kl[KT][DH + 8];        // K tile, bf16 row-major
  __shared__ __align__(16) __bf16 Vt[DH][KT + 8];        // V tile, bf16 transposed
  __shared__ __align__(16) __bf16 Pl[8][16][KT + 8];     // per-wave P relayout

  const int tid  = threadIdx.x;
  const int wave = tid >> 5;
  const int lane = tid & 31;
  const int ln   = lane & 15;   // row/col within 16
  const int g    = lane >> 4;   // half-wave group

  const int qt = blockIdx.x & 15;
  const int h  = (blockIdx.x >> 4) & 31;
  const int b  = blockIdx.x >> 9;
  const int hk = h >> 2;        // GQA: 4 query heads per kv head

  const int   q_base = qt * BM;
  const int   q_row0 = q_base + wave * 16;
  const float scale  = 0.08838834764831845f;  // 1/sqrt(128)

  // ---- load Q fragments (A layout: lane=row, interleaved K halves) ----
  Frag16 qf[4];
  {
    const float* qrow = q + (((size_t)b * S_LEN + (q_row0 + ln)) * HQ + h) * DH;
#pragma unroll
    for (int dsi = 0; dsi < 4; ++dsi) {
      const int d0 = dsi * 32 + g * 8;
      qf[dsi].h8[0] = cvt_f32x8(*(const float4*)(qrow + d0),
                                *(const float4*)(qrow + d0 + 4));
      qf[dsi].h8[1] = cvt_f32x8(*(const float4*)(qrow + d0 + 16),
                                *(const float4*)(qrow + d0 + 20));
    }
  }

  v8f  acc[8];
  float mrow[8], lrow[8], alpha[8];
#pragma unroll
  for (int t = 0; t < 8; ++t)
#pragma unroll
    for (int r = 0; r < 8; ++r) acc[t][r] = 0.0f;
#pragma unroll
  for (int r = 0; r < 8; ++r) { mrow[r] = -3.0e38f; lrow[r] = 0.0f; }

  const int ntiles  = (q_base + BM) / KT;   // causal: keys up to block's last row
  const int my_last = q_row0 + 15;          // wave-level causal cutoff

  // per-thread chunk for async prefetch: 8 x b128 per tile (K + V)
  // chunk c: key = c>>5, d4 = (c&31)*4
  // prologue: prefetch tile 0
#pragma unroll
  for (int i = 0; i < 4; ++i) {
    const int c   = tid + i * 256;
    const int key = c >> 5;
    const int d4  = (c & 31) * 4;
    const size_t base = (((size_t)b * S_LEN + key) * HKV_N + hk) * DH + d4;
    async_ld16(&Kf[key][d4], kg + base);
    async_ld16(&Vf[key][d4], vg + base);
  }

  for (int j = 0; j < ntiles; ++j) {
    const int key0 = j * KT;

    // ---- tile j landed in Kf/Vf (own wave) + all waves (barrier) ----
    wait_async0();
    __syncthreads();

    // ---- LDS->LDS convert pass: Kf->Kl (bf16), Vf->Vt (bf16, transposed) --
    {
      const int key = tid & 31;        // lanes spread keys -> conflict-free
      const int d0  = (tid >> 5) * 16; // 16 floats per thread per buffer
      const float4 k0 = *(const float4*)&Kf[key][d0];
      const float4 k1 = *(const float4*)&Kf[key][d0 + 4];
      const float4 k2 = *(const float4*)&Kf[key][d0 + 8];
      const float4 k3 = *(const float4*)&Kf[key][d0 + 12];
      *(v8bf*)&Kl[key][d0]     = cvt_f32x8(k0, k1);
      *(v8bf*)&Kl[key][d0 + 8] = cvt_f32x8(k2, k3);

      const float4 v0 = *(const float4*)&Vf[key][d0];
      const float4 v1 = *(const float4*)&Vf[key][d0 + 4];
      const float4 v2 = *(const float4*)&Vf[key][d0 + 8];
      const float4 v3 = *(const float4*)&Vf[key][d0 + 12];
      Vt[d0 +  0][key] = (__bf16)v0.x; Vt[d0 +  1][key] = (__bf16)v0.y;
      Vt[d0 +  2][key] = (__bf16)v0.z; Vt[d0 +  3][key] = (__bf16)v0.w;
      Vt[d0 +  4][key] = (__bf16)v1.x; Vt[d0 +  5][key] = (__bf16)v1.y;
      Vt[d0 +  6][key] = (__bf16)v1.z; Vt[d0 +  7][key] = (__bf16)v1.w;
      Vt[d0 +  8][key] = (__bf16)v2.x; Vt[d0 +  9][key] = (__bf16)v2.y;
      Vt[d0 + 10][key] = (__bf16)v2.z; Vt[d0 + 11][key] = (__bf16)v2.w;
      Vt[d0 + 12][key] = (__bf16)v3.x; Vt[d0 + 13][key] = (__bf16)v3.y;
      Vt[d0 + 14][key] = (__bf16)v3.z; Vt[d0 + 15][key] = (__bf16)v3.w;
    }
    __syncthreads();   // Kl/Vt ready; Kf/Vf fully consumed

    // ---- issue async prefetch of tile j+1 (overlaps with compute) ----
    if (j + 1 < ntiles) {
      const int nk0 = key0 + KT;
#pragma unroll
      for (int i = 0; i < 4; ++i) {
        const int c   = tid + i * 256;
        const int key = c >> 5;
        const int d4  = (c & 31) * 4;
        const size_t base =
            (((size_t)b * S_LEN + (nk0 + key)) * HKV_N + hk) * DH + d4;
        async_ld16(&Kf[key][d4], kg + base);
        async_ld16(&Vf[key][d4], vg + base);
      }
    }

    if (key0 <= my_last) {                 // wave-uniform causal skip
      // ---- S = Q @ K^T : two 16x16 score tiles, 4 bf16-WMMA K-steps each --
      v8f s0, s1;
#pragma unroll
      for (int r = 0; r < 8; ++r) { s0[r] = 0.0f; s1[r] = 0.0f; }
#pragma unroll
      for (int dsi = 0; dsi < 4; ++dsi) {
        Frag16 kb0, kb1;
        const v8bf* p0 = (const v8bf*)&Kl[ln][dsi * 32 + g * 8];
        const v8bf* p1 = (const v8bf*)&Kl[16 + ln][dsi * 32 + g * 8];
        kb0.h8[0] = p0[0]; kb0.h8[1] = p0[2];
        kb1.h8[0] = p1[0]; kb1.h8[1] = p1[2];
        s0 = __builtin_amdgcn_wmma_f32_16x16x32_bf16(
                 false, qf[dsi].v, false, kb0.v, (short)0, s0, false, false);
        s1 = __builtin_amdgcn_wmma_f32_16x16x32_bf16(
                 false, qf[dsi].v, false, kb1.v, (short)0, s1, false, false);
      }

      // ---- scale + causal mask (diagonal tiles only) ----
      const bool needmask = (key0 + KT - 1) > q_row0;
#pragma unroll
      for (int r = 0; r < 8; ++r) {
        const int rowg = q_row0 + g * 8 + r;   // C-layout: M = 8*g + r
        float a0 = s0[r] * scale;
        float a1 = s1[r] * scale;
        if (needmask) {
          if (key0 + ln > rowg)      a0 = -1.0e30f;
          if (key0 + 16 + ln > rowg) a1 = -1.0e30f;
        }
        s0[r] = a0; s1[r] = a1;
      }

      // ---- online softmax (f32 stats, replicated across 16-lane groups) --
#pragma unroll
      for (int r = 0; r < 8; ++r) {
        const float mx   = red_max16(fmaxf(s0[r], s1[r]));
        const float mnew = fmaxf(mrow[r], mx);
        const float p0 = __expf(s0[r] - mnew);
        const float p1 = __expf(s1[r] - mnew);
        alpha[r] = __expf(mrow[r] - mnew);
        mrow[r]  = mnew;
        lrow[r]  = lrow[r] * alpha[r] + red_sum16(p0 + p1);
        s0[r] = p0; s1[r] = p1;
      }
#pragma unroll
      for (int t = 0; t < 8; ++t)
#pragma unroll
        for (int r = 0; r < 8; ++r) acc[t][r] *= alpha[r];

      // ---- relayout P: C-matrix -> A-fragment via per-wave LDS ----
#pragma unroll
      for (int r = 0; r < 8; ++r) {
        Pl[wave][g * 8 + r][ln]      = (__bf16)s0[r];
        Pl[wave][g * 8 + r][16 + ln] = (__bf16)s1[r];
      }
      Frag16 pf;
      pf.h8[0] = *(const v8bf*)&Pl[wave][ln][g * 8];
      pf.h8[1] = *(const v8bf*)&Pl[wave][ln][16 + g * 8];

      // ---- O += P @ V : 8 d-tiles, K=32 keys per WMMA ----
#pragma unroll
      for (int t = 0; t < 8; ++t) {
        Frag16 vf;
        const v8bf* vp = (const v8bf*)&Vt[t * 16 + ln][g * 8];
        vf.h8[0] = vp[0]; vf.h8[1] = vp[2];
        acc[t] = __builtin_amdgcn_wmma_f32_16x16x32_bf16(
                     false, pf.v, false, vf.v, (short)0, acc[t], false, false);
      }
    }
  }

  // ---- epilogue: normalize and store O ----
#pragma unroll
  for (int r = 0; r < 8; ++r) {
    const int   srow = q_row0 + g * 8 + r;
    const float inv  = 1.0f / lrow[r];
    float* orow = out + (((size_t)b * S_LEN + srow) * HQ + h) * DH;
#pragma unroll
    for (int t = 0; t < 8; ++t) orow[t * 16 + ln] = acc[t][r] * inv;
  }
}

// ---------------- launch ----------------
extern "C" void kernel_launch(void* const* d_in, const int* in_sizes, int n_in,
                              void* d_out, int out_size, void* d_ws, size_t ws_size,
                              hipStream_t stream) {
  const float* q    = (const float*)d_in[0];
  const float* k    = (const float*)d_in[1];
  const float* v    = (const float*)d_in[2];
  const int*   slot = (const int*)d_in[5];

  float* out = (float*)d_out;                                   // [B,S,H,D]
  float* kc  = out + (size_t)NB * S_LEN * HQ * DH;              // [B*S, HKV*D]
  float* vc  = kc  + (size_t)NB * S_LEN * HKV_N * DH;

  kv_store_kernel<<<NB * S_LEN, 256, 0, stream>>>(k, v, slot, kc, vc);
  fa_fwd<<<NB * HQ * (S_LEN / BM), 256, 0, stream>>>(q, k, v, out);
}
